// VQE_28638841930032
// MI455X (gfx1250) — matrix operand order repaired and checked by reading, not compiled, not run on hardware
//
#include <hip/hip_runtime.h>

// ---------------------------------------------------------------------------
// CDNA5 WMMA (wave32): D(16x16 f32) = A(16x32 bf16) x B(32x16 bf16) + C
// ---------------------------------------------------------------------------
typedef __bf16       v16bf __attribute__((ext_vector_type(16)));
typedef float        v8f   __attribute__((ext_vector_type(8)));
typedef unsigned int v4u   __attribute__((ext_vector_type(4)));

union FragAB { v4u u[2]; v16bf v; };     // 32 bytes = 16 bf16 per lane
union Pack8  { v4u u; unsigned short h[8]; };

static __device__ __forceinline__ unsigned short f2bf(float f) {
  union { float f; unsigned int u; } c; c.f = f;
  unsigned int u = c.u;
  return (unsigned short)((u + 0x7FFFu + ((u >> 16) & 1u)) >> 16); // RNE
}
static __device__ __forceinline__ float bf2f(unsigned short h) {
  union { unsigned int u; float f; } c; c.u = ((unsigned int)h) << 16;
  return c.f;
}

// ---------------------------------------------------------------------------
// Zero the 1-pixel halo ring of a padded NHWC buffer (pitch Wp, height Hp).
// ---------------------------------------------------------------------------
__global__ void halo_zero_k(unsigned short* __restrict__ buf, int Hp, int Wp,
                            int cs, int total) {
  int idx = blockIdx.x * blockDim.x + threadIdx.x;
  if (idx >= total) return;
  int halo = 2 * Wp + 2 * (Hp - 2);
  int b = idx / halo, t = idx % halo;
  int y, x;
  if (t < Wp)           { y = 0;      x = t; }
  else if (t < 2 * Wp)  { y = Hp - 1; x = t - Wp; }
  else { int u = t - 2 * Wp; y = 1 + (u >> 1); x = (u & 1) ? (Wp - 1) : 0; }
  v4u z = {0u, 0u, 0u, 0u};
  v4u* p = (v4u*)(buf + ((size_t)(b * Hp + y) * Wp + x) * cs);
  for (int i = 0; i < cs / 8; ++i) p[i] = z;
}

// ---------------------------------------------------------------------------
// Pack the 13-channel input concat into padded NHWC bf16 (32-channel).
// ---------------------------------------------------------------------------
__global__ void pack_input_k(const float* __restrict__ xr, const float* __restrict__ xn,
                             const float* __restrict__ xf, const float* __restrict__ mn,
                             const float* __restrict__ mf, unsigned short* __restrict__ out,
                             int H, int W, int Hp, int Wp) {
  int idx = blockIdx.x * blockDim.x + threadIdx.x;   // b*H*W + s
  int HW = H * W;
  int s = idx % HW, b = idx / HW;
  int y = s / W, x = s % W;
  unsigned short* o = out + ((size_t)(b * Hp + y + 1) * Wp + x + 1) * 32;
  #pragma unroll
  for (int c = 0; c < 3; ++c) {
    o[c]     = f2bf(xr[(size_t)(b * 3 + c) * HW + s]);
    o[3 + c] = f2bf(xn[(size_t)(b * 3 + c) * HW + s]);
    o[6 + c] = f2bf(xf[(size_t)(b * 3 + c) * HW + s]);
  }
  #pragma unroll
  for (int c = 0; c < 2; ++c) {
    o[9 + c]  = f2bf(mn[(size_t)(b * 2 + c) * HW + s]);
    o[11 + c] = f2bf(mf[(size_t)(b * 2 + c) * HW + s]);
  }
  #pragma unroll
  for (int c = 13; c < 32; ++c) o[c] = 0;
}

// ---------------------------------------------------------------------------
// Pack OIHW 3x3 fp32 weights into bf16 A-fragment order:
// flat = ((tile*9 + tap)*nchunks + chunk)*512 + lane*16 + j
// lane<16 -> M=lane, K=j(0..15) ; lane>=16 -> M=lane-16, K=16+j(16..31)
// ---------------------------------------------------------------------------
__global__ void pack_w_conv3_k(const float* __restrict__ w, unsigned short* __restrict__ dst,
                               int Cout, int Cin, int nchunks, int total) {
  int idx = blockIdx.x * blockDim.x + threadIdx.x;
  if (idx >= total) return;
  int j    = idx & 15;
  int lane = (idx >> 4) & 31;
  int rest = idx >> 9;
  int chunk = rest % nchunks; rest /= nchunks;
  int tap   = rest % 9;
  int tile  = rest / 9;
  int half = lane >> 4;
  int m    = tile * 16 + (lane & 15);
  int K    = j + 16 * half;
  int cin  = chunk * 32 + K;
  float v = 0.f;
  if (m < Cout && cin < Cin) v = w[(size_t)(m * Cin + cin) * 9 + tap];
  dst[idx] = f2bf(v);
}

// Transpose-conv weights: src shape (Cin, Cout, 4, 4); 16 taps.
__global__ void pack_w_convt_k(const float* __restrict__ w, unsigned short* __restrict__ dst,
                               int Cout, int Cin, int nchunks, int total) {
  int idx = blockIdx.x * blockDim.x + threadIdx.x;
  if (idx >= total) return;
  int j    = idx & 15;
  int lane = (idx >> 4) & 31;
  int rest = idx >> 9;
  int chunk = rest % nchunks; rest /= nchunks;
  int tap   = rest & 15;
  int tile  = rest >> 4;
  int ky = tap >> 2, kx = tap & 3;
  int half = lane >> 4;
  int m    = tile * 16 + (lane & 15);
  int K    = j + 16 * half;
  int cin  = chunk * 32 + K;
  float v = 0.f;
  if (m < Cout && cin < Cin)
    v = w[(((size_t)cin * Cout + m) * 4 + ky) * 4 + kx];
  dst[idx] = f2bf(v);
}

// ---------------------------------------------------------------------------
// Implicit-GEMM 3x3 conv (pad=1) on halo-padded NHWC bf16 buffers.
// One wave = 16 Cout x 16 x-positions x ROWS output rows; each A (weight)
// fragment feeds ROWS WMMAs. No boundary predication (halo is zero).
// act: 0=none, 1=relu, 2=leaky(0.01), 3=final fp32 NCHW store of x_rec+res
// ---------------------------------------------------------------------------
template<int NCH, int ROWS>
__global__ void conv3x3_wmma_k(const unsigned short* __restrict__ xin,
                               const unsigned short* __restrict__ wpack,
                               const float* __restrict__ bias,
                               unsigned short* __restrict__ out,
                               int Hp_in, int Wp_in, int in_cs, int in_coff,
                               int Hout, int Wout, int Wp_out, int stride,
                               int out_cs, int out_coff, int Cout, int act,
                               const float* __restrict__ xrec, float* __restrict__ outf) {
  int lane = threadIdx.x & 31;
  int tile = threadIdx.x >> 5;
  int n = lane & 15, half = lane >> 4;
  int x0  = blockIdx.x << 4;
  int yo0 = blockIdx.y * ROWS;
  int b   = blockIdx.z;

  v8f acc[ROWS];
  #pragma unroll
  for (int r = 0; r < ROWS; ++r) acc[r] = (v8f){0.f,0.f,0.f,0.f,0.f,0.f,0.f,0.f};

  const unsigned short* wb = wpack + (size_t)tile * 9 * NCH * 512 + (lane << 4);
  int xb = (x0 + n) * stride;            // padded col for kx=0 (logical -1 + halo 1)
  size_t rowbase[ROWS];
  #pragma unroll
  for (int r = 0; r < ROWS; ++r)
    rowbase[r] = ((size_t)(b * Hp_in + (yo0 + r) * stride) * Wp_in + xb) * in_cs
               + in_coff + half * 16;

  #pragma unroll
  for (int tap = 0; tap < 9; ++tap) {
    size_t toff = ((size_t)(tap / 3) * Wp_in + (tap % 3)) * in_cs;
    #pragma unroll
    for (int c = 0; c < NCH; ++c) {
      FragAB a;
      const v4u* ap = (const v4u*)(wb + (tap * NCH + c) * 512);
      a.u[0] = ap[0]; a.u[1] = ap[1];
      #pragma unroll
      for (int r = 0; r < ROWS; ++r) {
        const v4u* q = (const v4u*)(xin + rowbase[r] + toff + c * 32);
        FragAB bb;
        bb.u[0] = q[0]; bb.u[1] = q[1];
        acc[r] = __builtin_amdgcn_wmma_f32_16x16x32_bf16(false, a.v, false, bb.v,
                                                         (short)0, acc[r], false, false);
      }
    }
  }

  int co0 = tile * 16 + half * 8;
  #pragma unroll
  for (int r = 0; r < ROWS; ++r) {
    int yo = yo0 + r;
    float rv[8];
    #pragma unroll
    for (int i = 0; i < 8; ++i) {
      int co = co0 + i;
      float v = acc[r][i] + ((co < Cout) ? bias[co] : 0.f);
      if (act == 1) v = v > 0.f ? v : 0.f;
      else if (act == 2) v = v > 0.f ? v : 0.01f * v;
      rv[i] = v;
    }
    if (act != 3) {
      unsigned short* op = out + ((size_t)(b * (Hout + 2) + yo + 1) * Wp_out + x0 + n + 1) * out_cs
                         + out_coff + co0;
      if (co0 + 8 <= Cout) {
        Pack8 p;
        #pragma unroll
        for (int i = 0; i < 8; ++i) p.h[i] = f2bf(rv[i]);
        *(v4u*)op = p.u;
      } else {
        #pragma unroll
        for (int i = 0; i < 8; ++i) if (co0 + i < Cout) op[i] = f2bf(rv[i]);
      }
    } else {
      int x = x0 + n;
      #pragma unroll
      for (int i = 0; i < 8; ++i) {
        int co = co0 + i;
        if (co < Cout) {
          size_t o = ((size_t)(b * Cout + co) * Hout + yo) * Wout + x;
          outf[o] = xrec[o] + rv[i];
        }
      }
    }
  }
}

// ---------------------------------------------------------------------------
// 4x4 stride-2 transpose conv (pad 1), by output parity: 4 four-tap stride-1
// convs. Inputs/outputs halo-padded NHWC bf16, relu.
// out[co,y,x] = sum w[ci,co,ky,kx] * in[ci,(y+1-ky)/2,(x+1-kx)/2]
// ---------------------------------------------------------------------------
__global__ void convt4x4_wmma_k(const unsigned short* __restrict__ xin,
                                const unsigned short* __restrict__ wpack,
                                const float* __restrict__ bias,
                                unsigned short* __restrict__ out,
                                int Hin, int Win, int in_cs,
                                int Wp_out, int out_cs, int out_coff) {
  const int NCH = 2;
  int lane = threadIdx.x & 31;
  int tile = threadIdx.x >> 5;
  int n = lane & 15, half = lane >> 4;
  int xo0 = blockIdx.x << 4;
  int yo  = blockIdx.y;
  int b   = blockIdx.z;
  int Wp_in = Win + 2, Hp_in = Hin + 2;

  const int kk[2][2] = {{1, 3}, {0, 2}};   // kernel taps per parity
  const int dd[2][2] = {{0, -1}, {1, 0}};  // input row/col offset per parity/tap

  const unsigned short* wb = wpack + (size_t)tile * 16 * NCH * 512 + (lane << 4);

  #pragma unroll
  for (int py = 0; py < 2; ++py) {
    #pragma unroll
    for (int px = 0; px < 2; ++px) {
      v8f acc = {0.f, 0.f, 0.f, 0.f, 0.f, 0.f, 0.f, 0.f};
      #pragma unroll
      for (int jy = 0; jy < 2; ++jy) {
        int yi = yo + dd[py][jy] + 1;            // padded row
        int ky = kk[py][jy];
        #pragma unroll
        for (int jx = 0; jx < 2; ++jx) {
          int xi = xo0 + n + dd[px][jx] + 1;     // padded col
          int kx = kk[px][jx];
          int tap = ky * 4 + kx;
          #pragma unroll
          for (int c = 0; c < NCH; ++c) {
            FragAB a, bb;
            const v4u* ap = (const v4u*)(wb + (tap * NCH + c) * 512);
            a.u[0] = ap[0]; a.u[1] = ap[1];
            const v4u* q = (const v4u*)(xin + ((size_t)(b * Hp_in + yi) * Wp_in + xi) * in_cs
                                            + half * 16 + c * 32);
            bb.u[0] = q[0]; bb.u[1] = q[1];
            acc = __builtin_amdgcn_wmma_f32_16x16x32_bf16(false, a.v, false, bb.v,
                                                          (short)0, acc, false, false);
          }
        }
      }
      int co0 = tile * 16 + half * 8;
      Pack8 p;
      #pragma unroll
      for (int i = 0; i < 8; ++i) {
        float v = acc[i] + bias[co0 + i];
        p.h[i] = f2bf(v > 0.f ? v : 0.f);
      }
      int y = 2 * yo + py, x = 2 * (xo0 + n) + px;
      unsigned short* op = out + ((size_t)(b * (2 * Hin + 2) + y + 1) * Wp_out + x + 1) * out_cs
                         + out_coff + co0;
      *(v4u*)op = p.u;
    }
  }
}

// ---------------------------------------------------------------------------
// Modulated deformable conv (dg=2, k=3, pad=1) fused with offset/mask
// post-processing (10*tanh + motion[::-1], sigmoid). One thread per pixel.
// Weights (64,6,3,3) staged in LDS. mi / feat are halo-padded NHWC bf16.
// ---------------------------------------------------------------------------
__global__ void deform_conv_k(const unsigned short* __restrict__ mi,
                              const float* __restrict__ xn, const float* __restrict__ xf,
                              const float* __restrict__ mn, const float* __restrict__ mf,
                              const float* __restrict__ w, const float* __restrict__ bias,
                              unsigned short* __restrict__ feat, int H, int W) {
  __shared__ float wl[64 * 54];
  for (int i = threadIdx.x; i < 64 * 54; i += blockDim.x) wl[i] = w[i];
  __syncthreads();

  int Hp = H + 2, Wp = W + 2;
  int x = blockIdx.x * blockDim.x + threadIdx.x;
  int y = blockIdx.y, b = blockIdx.z;
  int hw = H * W, s = y * W + x;
  const unsigned short* mp = mi + ((size_t)(b * Hp + y + 1) * Wp + x + 1) * 64;

  float v[54];
  for (int g = 0; g < 2; ++g) {
    const float* img = g ? xf : xn;     // (B,3,H,W)
    const float* mo  = g ? mf : mn;     // (B,2,H,W)
    float m0 = mo[(size_t)(b * 2 + 0) * hw + s];
    float m1 = mo[(size_t)(b * 2 + 1) * hw + s];
    for (int k = 0; k < 9; ++k) {
      float dy  = 10.f * tanhf(bf2f(mp[g * 27 + 2 * k]))     + m1;
      float dx  = 10.f * tanhf(bf2f(mp[g * 27 + 2 * k + 1])) + m0;
      float msk = 1.f / (1.f + expf(-bf2f(mp[g * 27 + 18 + k])));
      float py = (float)y + (float)(k / 3 - 1) + dy;
      float px = (float)x + (float)(k % 3 - 1) + dx;
      float fy = floorf(py), fx = floorf(px);
      float wy = py - fy, wx = px - fx;
      int y0 = (int)fy, x0i = (int)fx;
      float w00 = (1.f - wy) * (1.f - wx), w01 = (1.f - wy) * wx;
      float w10 = wy * (1.f - wx),         w11 = wy * wx;
      #pragma unroll
      for (int c = 0; c < 3; ++c) {
        const float* ip = img + (size_t)(b * 3 + c) * hw;
        float s00 = (y0 >= 0     && y0 < H     && x0i >= 0     && x0i < W)     ? ip[y0 * W + x0i]           : 0.f;
        float s01 = (y0 >= 0     && y0 < H     && x0i + 1 >= 0 && x0i + 1 < W) ? ip[y0 * W + x0i + 1]       : 0.f;
        float s10 = (y0 + 1 >= 0 && y0 + 1 < H && x0i >= 0     && x0i < W)     ? ip[(y0 + 1) * W + x0i]     : 0.f;
        float s11 = (y0 + 1 >= 0 && y0 + 1 < H && x0i + 1 >= 0 && x0i + 1 < W) ? ip[(y0 + 1) * W + x0i + 1] : 0.f;
        v[(g * 3 + c) * 9 + k] = (s00 * w00 + s01 * w01 + s10 * w10 + s11 * w11) * msk;
      }
    }
  }

  unsigned short* fp = feat + ((size_t)(b * Hp + y + 1) * Wp + x + 1) * 64;
  for (int o = 0; o < 64; ++o) {
    const float* wr = wl + o * 54;
    float acc = bias[o];
    #pragma unroll 9
    for (int i = 0; i < 54; ++i) acc += v[i] * wr[i];
    fp[o] = f2bf(acc);
  }
}

// ---------------------------------------------------------------------------
// Host-side launch
// ---------------------------------------------------------------------------
extern "C" void kernel_launch(void* const* d_in, const int* in_sizes, int n_in,
                              void* d_out, int out_size, void* d_ws, size_t ws_size,
                              hipStream_t stream) {
  (void)in_sizes; (void)n_in; (void)out_size; (void)ws_size;
  const int B = 4, H = 384, W = 384, HW = H * W;
  const int Hh = 192, Wh = 192;
  const int Hp = H + 2, Wp = W + 2;      // padded full-res
  const int Hph = Hh + 2, Wph = Wh + 2;  // padded half-res

  const float* x_rec = (const float*)d_in[0];
  const float* x_n   = (const float*)d_in[1];
  const float* x_f   = (const float*)d_in[2];
  const float* m_n   = (const float*)d_in[3];
  const float* m_f   = (const float*)d_in[4];
  const float* uw_in = (const float*)d_in[5];  const float* ub_in = (const float*)d_in[6];
  const float* uw_t1 = (const float*)d_in[7];  const float* ub_t1 = (const float*)d_in[8];
  const float* uw_t2 = (const float*)d_in[9];  const float* ub_t2 = (const float*)d_in[10];
  const float* uw_tt = (const float*)d_in[11]; const float* ub_tt = (const float*)d_in[12];
  const float* uw_out= (const float*)d_in[13]; const float* ub_out= (const float*)d_in[14];
  const float* dw    = (const float*)d_in[15]; const float* db    = (const float*)d_in[16];
  const float* qw1   = (const float*)d_in[17]; const float* qb1   = (const float*)d_in[18];
  const float* qw2   = (const float*)d_in[19]; const float* qb2   = (const float*)d_in[20];
  const float* qw3   = (const float*)d_in[21]; const float* qb3   = (const float*)d_in[22];

  char* ws = (char*)d_ws;
  size_t off = 0;
  auto alloc = [&](size_t bytes) -> char* {
    char* p = ws + off; off += (bytes + 255) & ~(size_t)255; return p;
  };
  unsigned short* IN0 = (unsigned short*)alloc((size_t)B * Hp * Wp * 32 * 2);
  unsigned short* CAT = (unsigned short*)alloc((size_t)B * Hp * Wp * 128 * 2);
  unsigned short* T1  = (unsigned short*)alloc((size_t)B * Hph * Wph * 64 * 2);
  unsigned short* T2  = (unsigned short*)alloc((size_t)B * Hph * Wph * 64 * 2);
  unsigned short* MI  = (unsigned short*)alloc((size_t)B * Hp * Wp * 64 * 2);
  unsigned short* H1  = (unsigned short*)alloc((size_t)B * Hp * Wp * 64 * 2);
  unsigned short* FEAT = CAT;  // concat buffer dead after uw_out conv
  unsigned short* H2   = MI;   // mi dead after deform conv
  unsigned short* PWIN = (unsigned short*)alloc((size_t)4 * 9 * 1 * 512 * 2);
  unsigned short* PWT1 = (unsigned short*)alloc((size_t)4 * 9 * 2 * 512 * 2);
  unsigned short* PWT2 = (unsigned short*)alloc((size_t)4 * 9 * 2 * 512 * 2);
  unsigned short* PWTT = (unsigned short*)alloc((size_t)4 * 16 * 2 * 512 * 2);
  unsigned short* PWO  = (unsigned short*)alloc((size_t)4 * 9 * 4 * 512 * 2);
  unsigned short* PWQ1 = (unsigned short*)alloc((size_t)4 * 9 * 2 * 512 * 2);
  unsigned short* PWQ2 = (unsigned short*)alloc((size_t)4 * 9 * 2 * 512 * 2);
  unsigned short* PWQ3 = (unsigned short*)alloc((size_t)1 * 9 * 2 * 512 * 2);

  auto halo = [&](unsigned short* buf, int hp, int wp, int cs) {
    int total = B * (2 * wp + 2 * (hp - 2));
    halo_zero_k<<<dim3((total + 255) / 256), dim3(256), 0, stream>>>(buf, hp, wp, cs, total);
  };

  // 0) zero halos of buffers consumed as conv inputs
  halo(IN0, Hp, Wp, 32);
  halo(CAT, Hp, Wp, 128);
  halo(T1,  Hph, Wph, 64);
  halo(T2,  Hph, Wph, 64);
  halo(H1,  Hp, Wp, 64);

  // 1) pack activations + weights to bf16
  pack_input_k<<<dim3((B * HW) / 256), dim3(256), 0, stream>>>(x_rec, x_n, x_f, m_n, m_f,
                                                               IN0, H, W, Hp, Wp);
  auto packc = [&](const float* w, unsigned short* dst, int Cout, int Cin, int ntiles, int nch) {
    int total = ntiles * 9 * nch * 512;
    pack_w_conv3_k<<<dim3((total + 255) / 256), dim3(256), 0, stream>>>(w, dst, Cout, Cin, nch, total);
  };
  packc(uw_in,  PWIN, 64, 13, 4, 1);
  packc(uw_t1,  PWT1, 64, 64, 4, 2);
  packc(uw_t2,  PWT2, 64, 64, 4, 2);
  packc(uw_out, PWO,  54, 128, 4, 4);
  packc(qw1,    PWQ1, 64, 64, 4, 2);
  packc(qw2,    PWQ2, 64, 64, 4, 2);
  packc(qw3,    PWQ3, 3,  64, 1, 2);
  {
    int total = 4 * 16 * 2 * 512;
    pack_w_convt_k<<<dim3((total + 255) / 256), dim3(256), 0, stream>>>(uw_tt, PWTT, 64, 64, 2, total);
  }

  const int ROWS = 4;
  // 2) U-Net
  // out1 = relu(conv(in0, uw_in)) -> CAT[ch 0..64)
  conv3x3_wmma_k<1, ROWS><<<dim3(W / 16, H / ROWS, B), dim3(128), 0, stream>>>(
      IN0, PWIN, ub_in, CAT, Hp, Wp, 32, 0, H, W, Wp, 1, 128, 0, 64, 1, nullptr, nullptr);
  // t = relu(conv_s2(out1, uw_t1))
  conv3x3_wmma_k<2, ROWS><<<dim3(Wh / 16, Hh / ROWS, B), dim3(128), 0, stream>>>(
      CAT, PWT1, ub_t1, T1, Hp, Wp, 128, 0, Hh, Wh, Wph, 2, 64, 0, 64, 1, nullptr, nullptr);
  // t = relu(conv(t, uw_t2))
  conv3x3_wmma_k<2, ROWS><<<dim3(Wh / 16, Hh / ROWS, B), dim3(128), 0, stream>>>(
      T1, PWT2, ub_t2, T2, Hph, Wph, 64, 0, Hh, Wh, Wph, 1, 64, 0, 64, 1, nullptr, nullptr);
  // out2 = relu(convT(t, uw_tt)) -> CAT[ch 64..128)
  convt4x4_wmma_k<<<dim3(Wh / 16, Hh, B), dim3(128), 0, stream>>>(
      T2, PWTT, ub_tt, CAT, Hh, Wh, 64, Wp, 128, 64);
  // mi = conv(cat, uw_out) (54 ch, no act)
  conv3x3_wmma_k<4, ROWS><<<dim3(W / 16, H / ROWS, B), dim3(128), 0, stream>>>(
      CAT, PWO, ub_out, MI, Hp, Wp, 128, 0, H, W, Wp, 1, 64, 0, 54, 0, nullptr, nullptr);

  // CAT is dead now; it becomes FEAT (cs=64) -> re-zero its halo in that layout
  halo(FEAT, Hp, Wp, 64);

  // 3) offsets/masks + deformable conv (fused)
  deform_conv_k<<<dim3(W / 128, H, B), dim3(128), 0, stream>>>(
      MI, x_n, x_f, m_n, m_f, dw, db, FEAT, H, W);

  // 4) QE head + residual
  conv3x3_wmma_k<2, ROWS><<<dim3(W / 16, H / ROWS, B), dim3(128), 0, stream>>>(
      FEAT, PWQ1, qb1, H1, Hp, Wp, 64, 0, H, W, Wp, 1, 64, 0, 64, 2, nullptr, nullptr);

  // MI is dead now; it becomes H2 (same cs=64 layout) -> zero halo
  halo(H2, Hp, Wp, 64);

  conv3x3_wmma_k<2, ROWS><<<dim3(W / 16, H / ROWS, B), dim3(128), 0, stream>>>(
      H1, PWQ2, qb2, H2, Hp, Wp, 64, 0, H, W, Wp, 1, 64, 0, 64, 2, nullptr, nullptr);
  conv3x3_wmma_k<2, ROWS><<<dim3(W / 16, H / ROWS, B), dim3(32), 0, stream>>>(
      H2, PWQ3, qb3, nullptr, Hp, Wp, 64, 0, H, W, Wp, 1, 0, 0, 3, 3, x_rec, (float*)d_out);
}
